// DynamicHyperConnection_47510928228705
// MI455X (gfx1250) — matrix operand order repaired and checked by reading, not compiled, not run on hardware
//
#include <hip/hip_runtime.h>
#include <hip/hip_bf16.h>
#include <cstdint>

// Problem constants (from reference): B=2,S=4096,D=2048,n=4
#define TOK   8192
#define DIM   2048
#define NEXP  4
#define DQ    512
#define NP    24     // n*n + n + n

typedef __attribute__((ext_vector_type(16))) __bf16 v16bf;
typedef __attribute__((ext_vector_type(8)))  float  v8f;

union BF16Frag { v16bf v; unsigned short u[16]; uint4 q[2]; };

__device__ __forceinline__ unsigned short f2bf(float f) {
    unsigned int u = __float_as_uint(f);
    unsigned int r = u + 0x7fffu + ((u >> 16) & 1u);   // round-to-nearest-even
    return (unsigned short)(r >> 16);
}

// A fragment (16x32 bf16, row-major source, row stride in elements):
// lane m = lane&15; VGPR0-3: K = 8*half + 0..7 ; VGPR4-7: K = 16 + 8*half + 0..7
__device__ __forceinline__ v16bf load_frag_a_g(const unsigned short* base, size_t stride,
                                               int lane) {
    const int m = lane & 15, half = lane >> 4;
    const unsigned short* p = base + (size_t)m * stride + 8 * half;
    BF16Frag f;
    f.q[0] = *(const uint4*)p;
    f.q[1] = *(const uint4*)(p + 16);
    return f.v;
}

// Same A layout but from an LDS tile (row stride 32).
__device__ __forceinline__ v16bf load_frag_a_lds(const unsigned short* a, int lane) {
    const int m = lane & 15, half = lane >> 4;
    const unsigned short* p = a + m * 32 + 8 * half;
    BF16Frag f;
    f.q[0] = *(const uint4*)p;
    f.q[1] = *(const uint4*)(p + 16);
    return f.v;
}

// B fragment (32x16 KxN bf16) from TRANSPOSED weights (row = N column, K contiguous):
// lane n = lane&15; VGPR0-3: K = 8*half + 0..7 ; VGPR4-7: K = 16 + 8*half + 0..7
__device__ __forceinline__ v16bf load_frag_b_g(const unsigned short* baseT, size_t stride,
                                               int lane) {
    const int nn = lane & 15, half = lane >> 4;
    const unsigned short* p = baseT + (size_t)nn * stride + 8 * half;
    BF16Frag f;
    f.q[0] = *(const uint4*)p;
    f.q[1] = *(const uint4*)(p + 16);
    return f.v;
}

// ---------------------------------------------------------------------------
// Kernel 0: f32 -> bf16 + transpose (dst[c][r] = bf16(src[r][c]))
// src: R x C f32, dst: C x R bf16. grid = (C/32, R/32), block = 256 (32x8)
// ---------------------------------------------------------------------------
__global__ __launch_bounds__(256)
void transpose_bf16_kernel(const float* __restrict__ src, unsigned short* __restrict__ dst,
                           int R, int C) {
    __shared__ unsigned short tile[32][33];
    const int r0 = blockIdx.y * 32, c0 = blockIdx.x * 32;
    const int tc = threadIdx.x & 31, tr = threadIdx.x >> 5;   // 32 x 8
#pragma unroll
    for (int i = 0; i < 32; i += 8)
        tile[tc][tr + i] = f2bf(src[(size_t)(r0 + tr + i) * C + c0 + tc]);
    __syncthreads();
#pragma unroll
    for (int i = 0; i < 32; i += 8)
        dst[(size_t)(c0 + tr + i) * R + r0 + tc] = tile[tr + i][tc];
}

// ---------------------------------------------------------------------------
// Kernel 1: per 16-token block:
//   x_proxy = mean_n H ; hmlp = relu(x@W1+b1) via bf16 WMMA.
//   Statically double-buffered: B frags for step k+32 are issued before the
//   barrier of step k so they fly during barrier + A staging.
//   params = hmlp@W2+b2 (in-LDS); h_in -> bf16 (H L2-hot).
// ---------------------------------------------------------------------------
__global__ __launch_bounds__(256)
void param_mlp_kernel(const float* __restrict__ H, const float* __restrict__ b1,
                      const float* __restrict__ W2, const float* __restrict__ b2,
                      const unsigned short* __restrict__ W1T,   // DQ x DIM bf16
                      float* __restrict__ params, unsigned short* __restrict__ hin_bf) {
    __shared__ __align__(16) unsigned short aLds[2][16 * 32];
    __shared__ __align__(16) float hmlp[16 * 512];
    __shared__ __align__(16) float pLds[16 * NP];

    const int tid  = threadIdx.x;
    const int lane = tid & 31;
    const int wave = tid >> 5;           // 0..7, each owns 64 N-columns
    const int t0   = blockIdx.x * 16;

    v8f acc[4] = {};
    v16bf bF0[4], bF1[4];

    auto loadB = [&](v16bf (&bF)[4], int kk) {
#pragma unroll
        for (int nt = 0; nt < 4; ++nt)
            bF[nt] =
                load_frag_b_g(W1T + (size_t)(wave * 64 + nt * 16) * DIM + kk, DIM, lane);
    };
    auto stageA = [&](unsigned short* aBuf, int kk) {
        const int ti = tid >> 4, ki = (tid & 15) * 2;
        const float4 h0 =
            *(const float4*)(H + ((size_t)(t0 + ti) * DIM + kk + ki) * NEXP);
        const float4 h1 =
            *(const float4*)(H + ((size_t)(t0 + ti) * DIM + kk + ki + 1) * NEXP);
        const unsigned int pk =
            (unsigned int)f2bf((h0.x + h0.y + h0.z + h0.w) * 0.25f) |
            ((unsigned int)f2bf((h1.x + h1.y + h1.z + h1.w) * 0.25f) << 16);
        *(unsigned int*)(aBuf + ti * 32 + ki) = pk;
    };
    auto mma = [&](const v16bf (&bF)[4], const v16bf afrag) {
#pragma unroll
        for (int nt = 0; nt < 4; ++nt)
            acc[nt] = __builtin_amdgcn_wmma_f32_16x16x32_bf16(
                false, afrag, false, bF[nt], (short)0, acc[nt], false, false);
    };

    loadB(bF0, 0);
    for (int k0 = 0; k0 < DIM; k0 += 64) {     // 32 iterations, buffers static
        // even step: LDS buf 0 / bF0 ; prefetch bF1
        stageA(aLds[0], k0);
        loadB(bF1, k0 + 32);
        __syncthreads();
        mma(bF0, load_frag_a_lds(aLds[0], lane));
        // odd step: LDS buf 1 / bF1 ; prefetch bF0
        stageA(aLds[1], k0 + 32);
        if (k0 + 64 < DIM) loadB(bF0, k0 + 64);
        __syncthreads();
        mma(bF1, load_frag_a_lds(aLds[1], lane));
    }

    // relu(acc + b1) -> hmlp LDS (C layout: N=lane&15, M=r+8*(lane>>4))
#pragma unroll
    for (int nt = 0; nt < 4; ++nt) {
        const int col = wave * 64 + nt * 16 + (lane & 15);
        const int mo  = (lane >> 4) * 8;
        const float bb = b1[col];
#pragma unroll
        for (int r = 0; r < 8; ++r) {
            const float v = acc[nt][r] + bb;
            hmlp[(mo + r) * 512 + col] = v > 0.f ? v : 0.f;
        }
    }
    __syncthreads();

    // GEMM2: 16 tokens x 24 params, K=512 (tiny, scalar FMA)
    for (int o = tid; o < 16 * NP; o += 256) {
        const int tl = o / NP, p = o % NP;
        float s = b2[p];
        for (int f = 0; f < DQ; ++f) s += hmlp[tl * 512 + f] * W2[f * NP + p];
        pLds[o] = s;
        params[(size_t)(t0 + tl) * NP + p] = s;
    }
    __syncthreads();

    // h_in = sum_n H * P_input -> bf16 workspace
    for (int idx = tid; idx < 16 * DIM; idx += 256) {
        const int tl = idx >> 11, d = idx & (DIM - 1);
        const float4 h4 =
            *(const float4*)(H + ((size_t)(t0 + tl) * DIM + d) * NEXP);
        const float* pin = &pLds[tl * NP + 16];
        const float v = h4.x * pin[0] + h4.y * pin[1] + h4.z * pin[2] + h4.w * pin[3];
        hin_bf[(size_t)(t0 + tl) * DIM + d] = f2bf(v);
    }
}

// ---------------------------------------------------------------------------
// Kernel 2: y = h_in @ Wm + bm via bf16 WMMA, all fragments straight from
// global (L2-resident). Statically double-buffered software pipeline:
// 8 b128-load-pairs of step k+32 fly while 16 WMMAs run on step k.
// Epilogue: out[t,e,m] = sum_n H[t,e,n]*P_width[n,m] + y*P_depth[m]
// grid: (TOK/64, DIM/512)
// ---------------------------------------------------------------------------
__global__ __launch_bounds__(256)
void out_gemm_kernel(const float* __restrict__ H, const float* __restrict__ bm,
                     const unsigned short* __restrict__ WmT,     // DIM x DIM bf16 (T)
                     const unsigned short* __restrict__ hin_bf,  // TOK x DIM bf16
                     const float* __restrict__ params, float* __restrict__ out) {
    __shared__ __align__(16) float pLds[64 * NP];

    const int tid  = threadIdx.x;
    const int lane = tid & 31;
    const int wave = tid >> 5;           // 0..7, each owns 64 N-columns
    const int t0   = blockIdx.x * 64;
    const int e0   = blockIdx.y * 512;

    for (int i = tid; i < 64 * NP; i += 256) pLds[i] = params[(size_t)t0 * NP + i];
    __syncthreads();

    v8f acc[4][4] = {};   // [M-tile][N-tile]
    v16bf aF0[4], bF0[4], aF1[4], bF1[4];

    auto loadStep = [&](v16bf (&aF)[4], v16bf (&bF)[4], int kk) {
#pragma unroll
        for (int mt = 0; mt < 4; ++mt)
            aF[mt] =
                load_frag_a_g(hin_bf + (size_t)(t0 + mt * 16) * DIM + kk, DIM, lane);
#pragma unroll
        for (int nt = 0; nt < 4; ++nt)
            bF[nt] = load_frag_b_g(
                WmT + (size_t)(e0 + wave * 64 + nt * 16) * DIM + kk, DIM, lane);
    };
    auto mmaStep = [&](const v16bf (&aF)[4], const v16bf (&bF)[4]) {
#pragma unroll
        for (int nt = 0; nt < 4; ++nt)
#pragma unroll
            for (int mt = 0; mt < 4; ++mt)
                acc[mt][nt] = __builtin_amdgcn_wmma_f32_16x16x32_bf16(
                    false, aF[mt], false, bF[nt], (short)0, acc[mt][nt], false, false);
    };

    loadStep(aF0, bF0, 0);
    for (int k0 = 0; k0 < DIM; k0 += 64) {     // 32 iterations, buffers static
        loadStep(aF1, bF1, k0 + 32);
        mmaStep(aF0, bF0);
        if (k0 + 64 < DIM) loadStep(aF0, bF0, k0 + 64);
        mmaStep(aF1, bF1);
    }

    // Epilogue: y = acc + bm[e]; out = H.P_width + y*P_depth (read H once)
#pragma unroll
    for (int mt = 0; mt < 4; ++mt) {
#pragma unroll
        for (int nt = 0; nt < 4; ++nt) {
            const int e  = e0 + wave * 64 + nt * 16 + (lane & 15);
            const int mo = (lane >> 4) * 8;
            const float bme = bm[e];
#pragma unroll
            for (int r = 0; r < 8; ++r) {
                const int tl = mt * 16 + mo + r;
                const float y = acc[mt][nt][r] + bme;
                const float4 h4 =
                    *(const float4*)(H + ((size_t)(t0 + tl) * DIM + e) * NEXP);
                const float* pw = &pLds[tl * NP];
                float4 o;
                o.x = h4.x*pw[0] + h4.y*pw[4] + h4.z*pw[8]  + h4.w*pw[12] + y*pw[20];
                o.y = h4.x*pw[1] + h4.y*pw[5] + h4.z*pw[9]  + h4.w*pw[13] + y*pw[21];
                o.z = h4.x*pw[2] + h4.y*pw[6] + h4.z*pw[10] + h4.w*pw[14] + y*pw[22];
                o.w = h4.x*pw[3] + h4.y*pw[7] + h4.z*pw[11] + h4.w*pw[15] + y*pw[23];
                *(float4*)(out + ((size_t)(t0 + tl) * DIM + e) * NEXP) = o;
            }
        }
    }
}

// ---------------------------------------------------------------------------
extern "C" void kernel_launch(void* const* d_in, const int* in_sizes, int n_in,
                              void* d_out, int out_size, void* d_ws, size_t ws_size,
                              hipStream_t stream) {
    (void)in_sizes; (void)n_in; (void)out_size; (void)ws_size;
    const float* H  = (const float*)d_in[0];
    const float* W1 = (const float*)d_in[1];
    const float* b1 = (const float*)d_in[2];
    const float* W2 = (const float*)d_in[3];
    const float* b2 = (const float*)d_in[4];
    const float* Wm = (const float*)d_in[5];
    const float* bm = (const float*)d_in[6];
    float* out = (float*)d_out;

    // Workspace layout (bytes):
    //   [0, 2MiB)        W1T bf16  (512 x 2048)
    //   [+, 8MiB)        WmT bf16  (2048 x 2048)
    //   [+, 768KiB)      params    (8192*24 f32)
    //   [+, 32MiB)       h_in bf16 (8192 x 2048)
    char* ws = (char*)d_ws;
    const size_t OFF_W1T    = 0;
    const size_t OFF_WMT    = OFF_W1T + (size_t)DIM * DQ * 2;
    const size_t OFF_PARAMS = OFF_WMT + (size_t)DIM * DIM * 2;
    const size_t OFF_HIN    = OFF_PARAMS + (size_t)TOK * NP * 4;
    unsigned short* W1T     = (unsigned short*)(ws + OFF_W1T);
    unsigned short* WmT     = (unsigned short*)(ws + OFF_WMT);
    float*          paramsW = (float*)(ws + OFF_PARAMS);
    unsigned short* hinBf   = (unsigned short*)(ws + OFF_HIN);

    // W1: 2048x512 -> W1T 512x2048 ; Wm: 2048x2048 -> WmT 2048x2048
    transpose_bf16_kernel<<<dim3(DQ / 32, DIM / 32), 256, 0, stream>>>(W1, W1T, DIM, DQ);
    transpose_bf16_kernel<<<dim3(DIM / 32, DIM / 32), 256, 0, stream>>>(Wm, WmT, DIM, DIM);

    param_mlp_kernel<<<TOK / 16, 256, 0, stream>>>(H, b1, W2, b2, W1T, paramsW, hinBf);

    out_gemm_kernel<<<dim3(TOK / 64, DIM / 512), 256, 0, stream>>>(H, bm, WmT, hinBf,
                                                                   paramsW, out);
}
